// SetAbstraction_89438398972560
// MI455X (gfx1250) — matrix operand order repaired and checked by reading, not compiled, not run on hardware
//
#include <hip/hip_runtime.h>

typedef __attribute__((ext_vector_type(16))) __bf16 v16bf;
typedef __attribute__((ext_vector_type(8)))  float  v8f;

#define BB   4
#define NN   4096
#define BN   (BB * NN)
#define CC   64
#define KNB  32          // max neighbors
#define H3C  128
#define RAD2 0.04f
#define CAP  256         // per-target candidate cap (expected ~137 in ball)

// ---- workspace layout ----
#define NB_BYTES  ((size_t)BN * KNB * 4)            // neighbor index lists (2 MB)
#define P1_FRAGS  12                                // 4 col-tiles x 3 k-chunks (K padded 67->96)
#define P2_FRAGS  8                                 // 4 x 2
#define P3_FRAGS  16                                // 8 x 2
#define P1_OFF    NB_BYTES
#define P2_OFF    (P1_OFF + (size_t)P1_FRAGS * 512 * 2)
#define P3_OFF    (P2_OFF + (size_t)P2_FRAGS * 512 * 2)

__device__ __forceinline__ __bf16 to_bf(float f) {
    unsigned u = __float_as_uint(f);
    unsigned r = (u + 0x7FFFu + ((u >> 16) & 1u)) >> 16;   // RNE
    unsigned short s = (unsigned short)r;
    return __builtin_bit_cast(__bf16, s);
}

// ============================================================================
// Kernel 1: convert + pack weights into WMMA B-fragment order (bf16).
// B-fragment (32x16 bf16 -> v16bf/lane): lane<16: N=lane, e0..7->K=ck*32+e, e8..15->K=ck*32+16+(e-8)
//                                        lane>=16: N=lane-16, same with K +8
// ============================================================================
__device__ void pack_one(const float* __restrict__ W, int Kdim, int Nout,
                         int NC, int NT, __bf16* __restrict__ out) {
    int total = NT * NC * 512;
    for (int idx = blockIdx.x * blockDim.x + threadIdx.x; idx < total;
         idx += gridDim.x * blockDim.x) {
        int e    = idx & 15;
        int lane = (idx >> 4) & 31;
        int f    = idx >> 9;
        int ck   = f % NC;
        int tt   = f / NC;
        int n    = tt * 16 + (lane & 15);
        int kk   = ck * 32 + ((lane & 16) ? 8 : 0) + (e & 7) + ((e & 8) ? 16 : 0);
        float v  = (kk < Kdim) ? W[(size_t)kk * Nout + n] : 0.f;
        out[idx] = to_bf(v);
    }
}

__global__ __launch_bounds__(256) void pack_weights(
    const float* __restrict__ W1, const float* __restrict__ W2,
    const float* __restrict__ W3, __bf16* p1, __bf16* p2, __bf16* p3) {
    pack_one(W1, 67, 64, 3, 4, p1);
    pack_one(W2, 64, 64, 2, 4, p2);
    pack_one(W3, 64, 128, 2, 8, p3);
}

// ============================================================================
// Kernel 2: radius-kNN. One wave32 per target point. Ballot-compacted append
// of in-ball candidates into per-wave LDS buffer, then 32 rounds of packed
// (d2,pos) min-reduction -> K nearest within ball (self d2=0 always first).
// ============================================================================
__global__ __launch_bounds__(256) void radius_knn(
    const float* __restrict__ pos, int* __restrict__ nb) {
    __shared__ float sD[8][CAP];
    __shared__ int   sI[8][CAP];

    const int w = threadIdx.x >> 5;
    const int l = threadIdx.x & 31;
    const int t = blockIdx.x * 8 + w;         // target row in [0, BN)
    const int b = t >> 12;                    // N == 4096
    const int i = t & (NN - 1);

    const float* pg = pos + (size_t)b * NN * 3;
    const float xi = pg[i * 3 + 0];
    const float yi = pg[i * 3 + 1];
    const float zi = pg[i * 3 + 2];

    int cnt = 0;
    for (int j0 = 0; j0 < NN; j0 += 32) {
        int j = j0 + l;
        float dx = pg[j * 3 + 0] - xi;
        float dy = pg[j * 3 + 1] - yi;
        float dz = pg[j * 3 + 2] - zi;
        float d2 = dx * dx + dy * dy + dz * dz;
        bool v = (d2 <= RAD2);
        unsigned mask = (unsigned)__ballot(v);
        int off = __popc(mask & ((1u << l) - 1u));
        int slot = cnt + off;
        if (v && slot < CAP) { sD[w][slot] = d2; sI[w][slot] = b * NN + j; }
        cnt += __popc(mask);
    }
    if (cnt > CAP) cnt = CAP;
    __syncthreads();

    for (int k = 0; k < KNB; ++k) {
        unsigned long long key = ~0ull;
        for (int c = l; c < cnt; c += 32) {
            unsigned long long kk =
                ((unsigned long long)__float_as_uint(sD[w][c]) << 32) | (unsigned)c;
            if (kk < key) key = kk;
        }
        #pragma unroll
        for (int s = 16; s > 0; s >>= 1) {
            unsigned long long o = __shfl_xor(key, s);
            if (o < key) key = o;
        }
        bool found = (unsigned)(key >> 32) < 0x7F800000u;   // finite d2
        if (l == 0) {
            int bp = (int)(key & 0xffffffffu);
            nb[(size_t)t * KNB + k] = found ? sI[w][bp] : -1;
            if (found) sD[w][bp] = __uint_as_float(0x7F800000u);  // remove
        }
        __syncthreads();
    }
}

// ============================================================================
// Kernel 3: per-edge MLP (bf16 WMMA, f32 accum) + masked max aggregation.
// One block (2 waves) per target; each wave runs 16 edges through
// [96->64->64->128] via v_wmma_f32_16x16x32_bf16 (12 + 8 + 16 = 36 WMMAs).
// A-fragment layout (16-bit A 16x32): lane<16: M=lane, e0..7->K=koff+e,
// e8..15->K=koff+16+(e-8), koff=ck*32 (+8 for lanes 16..31).
// D layout: lane holds col N=l&15, rows M=r (+8 for lanes 16..31).
// ============================================================================
__global__ __launch_bounds__(64) void edge_mlp(
    const float* __restrict__ x, const float* __restrict__ pos,
    const int* __restrict__ nb,
    const __bf16* __restrict__ p1, const __bf16* __restrict__ p2,
    const __bf16* __restrict__ p3,
    const float* __restrict__ b1, const float* __restrict__ b2,
    const float* __restrict__ b3,
    float* __restrict__ out) {
    __shared__ __bf16 sA[2][16][96];
    __shared__ __bf16 sH[2][16][64];
    __shared__ float  sOut[2][128];
    __shared__ int    sIdx[32];

    const int tid = threadIdx.x;
    const int w   = tid >> 5;
    const int l   = tid & 31;
    const int t   = blockIdx.x;               // target row == global point row

    if (tid < 32) sIdx[tid] = nb[(size_t)t * KNB + tid];
    __syncthreads();

    // ---- gather [x_j | pos_j - pos_i | 0-pad] into 16x96 bf16 tile ----
    {
        const int e    = l & 15;
        const int half = l >> 4;
        const int j    = sIdx[w * 16 + e];
        __bf16* arow = &sA[w][e][0];
        if (j >= 0) {
            const float* xr = x + (size_t)j * CC + half * 32;
            #pragma unroll
            for (int c = 0; c < 32; ++c) arow[half * 32 + c] = to_bf(xr[c]);
        } else {
            #pragma unroll
            for (int c = 0; c < 32; ++c) arow[half * 32 + c] = to_bf(0.f);
        }
        if (half == 0) {
            #pragma unroll
            for (int d = 0; d < 3; ++d) {
                float r = (j >= 0) ? (pos[(size_t)j * 3 + d] - pos[(size_t)t * 3 + d]) : 0.f;
                arow[64 + d] = to_bf(r);
            }
        } else {
            for (int c = 67; c < 96; ++c) arow[c] = to_bf(0.f);
        }
    }
    __syncthreads();

    const int hi  = (l & 16) ? 1 : 0;
    const int n16 = l & 15;
    const float NEGMAX = -3.4028234663852886e38f;

    // ---- layer 1: [16x96] @ [96x64], bias+ReLU -> sH ----
    v16bf a1[3];
    #pragma unroll
    for (int ck = 0; ck < 3; ++ck) {
        const __bf16* row = &sA[w][n16][0];
        const int koff = ck * 32 + hi * 8;
        #pragma unroll
        for (int e = 0; e < 8; ++e) { a1[ck][e] = row[koff + e]; a1[ck][e + 8] = row[koff + 16 + e]; }
    }
    #pragma unroll
    for (int t1 = 0; t1 < 4; ++t1) {
        v8f acc = {};
        #pragma unroll
        for (int ck = 0; ck < 3; ++ck) {
            v16bf bf = *(const v16bf*)(p1 + ((size_t)(t1 * 3 + ck) * 512 + l * 16));
            acc = __builtin_amdgcn_wmma_f32_16x16x32_bf16(false, a1[ck], false, bf,
                                                          (short)0, acc, false, false);
        }
        const float bias = b1[t1 * 16 + n16];
        #pragma unroll
        for (int r = 0; r < 8; ++r) {
            float v = acc[r] + bias; v = v > 0.f ? v : 0.f;
            sH[w][hi * 8 + r][t1 * 16 + n16] = to_bf(v);
        }
    }
    __syncthreads();

    // ---- layer 2: [16x64] @ [64x64], bias+ReLU -> sH (in place) ----
    v16bf a2[2];
    #pragma unroll
    for (int ck = 0; ck < 2; ++ck) {
        const __bf16* row = &sH[w][n16][0];
        const int koff = ck * 32 + hi * 8;
        #pragma unroll
        for (int e = 0; e < 8; ++e) { a2[ck][e] = row[koff + e]; a2[ck][e + 8] = row[koff + 16 + e]; }
    }
    __syncthreads();
    #pragma unroll
    for (int t2 = 0; t2 < 4; ++t2) {
        v8f acc = {};
        #pragma unroll
        for (int ck = 0; ck < 2; ++ck) {
            v16bf bf = *(const v16bf*)(p2 + ((size_t)(t2 * 2 + ck) * 512 + l * 16));
            acc = __builtin_amdgcn_wmma_f32_16x16x32_bf16(false, a2[ck], false, bf,
                                                          (short)0, acc, false, false);
        }
        const float bias = b2[t2 * 16 + n16];
        #pragma unroll
        for (int r = 0; r < 8; ++r) {
            float v = acc[r] + bias; v = v > 0.f ? v : 0.f;
            sH[w][hi * 8 + r][t2 * 16 + n16] = to_bf(v);
        }
    }
    __syncthreads();

    // ---- layer 3: [16x64] @ [64x128], bias+ReLU, masked max over edges ----
    v16bf a3[2];
    #pragma unroll
    for (int ck = 0; ck < 2; ++ck) {
        const __bf16* row = &sH[w][n16][0];
        const int koff = ck * 32 + hi * 8;
        #pragma unroll
        for (int e = 0; e < 8; ++e) { a3[ck][e] = row[koff + e]; a3[ck][e + 8] = row[koff + 16 + e]; }
    }
    #pragma unroll
    for (int t3 = 0; t3 < 8; ++t3) {
        v8f acc = {};
        #pragma unroll
        for (int ck = 0; ck < 2; ++ck) {
            v16bf bf = *(const v16bf*)(p3 + ((size_t)(t3 * 2 + ck) * 512 + l * 16));
            acc = __builtin_amdgcn_wmma_f32_16x16x32_bf16(false, a3[ck], false, bf,
                                                          (short)0, acc, false, false);
        }
        const float bias = b3[t3 * 16 + n16];
        float m = NEGMAX;
        #pragma unroll
        for (int r = 0; r < 8; ++r) {
            const int M = hi * 8 + r;
            const bool ok = sIdx[w * 16 + M] >= 0;
            float v = acc[r] + bias; v = v > 0.f ? v : 0.f;
            m = fmaxf(m, ok ? v : NEGMAX);
        }
        m = fmaxf(m, __shfl_xor(m, 16));     // combine rows 0-7 with 8-15
        if (hi == 0) sOut[w][t3 * 16 + n16] = m;
    }
    __syncthreads();

    out[(size_t)t * H3C + tid]      = fmaxf(sOut[0][tid],      sOut[1][tid]);
    out[(size_t)t * H3C + 64 + tid] = fmaxf(sOut[0][64 + tid], sOut[1][64 + tid]);
}

// ============================================================================
extern "C" void kernel_launch(void* const* d_in, const int* in_sizes, int n_in,
                              void* d_out, int out_size, void* d_ws, size_t ws_size,
                              hipStream_t stream) {
    const float* x    = (const float*)d_in[0];
    const float* pos  = (const float*)d_in[1];
    const int*   batch= (const int*)  d_in[2];
    const float* W1   = (const float*)d_in[3];
    const float* b1   = (const float*)d_in[4];
    const float* W2   = (const float*)d_in[5];
    const float* b2   = (const float*)d_in[6];
    const float* W3   = (const float*)d_in[7];
    const float* b3   = (const float*)d_in[8];
    float* out = (float*)d_out;

    char* ws = (char*)d_ws;
    int*    nb = (int*)ws;
    __bf16* p1 = (__bf16*)(ws + P1_OFF);
    __bf16* p2 = (__bf16*)(ws + P2_OFF);
    __bf16* p3 = (__bf16*)(ws + P3_OFF);

    pack_weights<<<32, 256, 0, stream>>>(W1, W2, W3, p1, p2, p3);
    radius_knn<<<BN / 8, 256, 0, stream>>>(pos, nb);
    edge_mlp<<<BN, 64, 0, stream>>>(x, pos, nb, p1, p2, p3, b1, b2, b3, out);

    // reference returns (out, pos, batch): append the passthroughs.
    hipMemcpyAsync((char*)d_out + (size_t)BN * H3C * 4, d_in[1],
                   (size_t)BN * 3 * 4, hipMemcpyDeviceToDevice, stream);
    hipMemcpyAsync((char*)d_out + (size_t)BN * H3C * 4 + (size_t)BN * 3 * 4, d_in[2],
                   (size_t)BN * 4, hipMemcpyDeviceToDevice, stream);
}
